// VesselDHNet_73985106641563
// MI455X (gfx1250) — compile-verified, matrix operand order checked by reading, not run on hardware
//
#include <hip/hip_runtime.h>
#include <math.h>

// ---- problem constants (from reference) ----
#define F_IN   27
#define NH1    8
#define NC1    8
#define F1     64      // NH1*NC1
#define NC2    2
#define NEGSL  0.2f

typedef __attribute__((ext_vector_type(16))) _Float16 v16h;
typedef __attribute__((ext_vector_type(8)))  float    v8f;

__device__ __forceinline__ float leakyr(float t) { return t > 0.f ? t : NEGSL * t; }

// float atomic max via signed-max / unsigned-min bit trick (always available)
__device__ __forceinline__ void atomicMaxF(float* addr, float val) {
  if (val >= 0.f)
    atomicMax((int*)addr, __float_as_int(val));
  else
    atomicMin((unsigned int*)addr, __float_as_uint(val));
}

__device__ __forceinline__ void edge_sd(const int* __restrict__ ei, int e, int E,
                                        int& s, int& d) {
  if (e < E) { s = ei[e]; d = ei[E + e]; }
  else       { s = e - E; d = e - E; }
}

// ---------------------------------------------------------------------------
// Kernel 1: h1[N,64] = x[N,27] @ W1[27,64] via v_wmma_f32_16x16x32_f16.
// One wave computes a 16-row x 64-col block (4 WMMAs, K padded 27->32).
// ---------------------------------------------------------------------------
__global__ __launch_bounds__(128)
void k_gemm1(const float* __restrict__ x, const float* __restrict__ W1,
             float* __restrict__ h1, int N) {
  const int wave = threadIdx.x >> 5;
  const int lane = threadIdx.x & 31;
  const int rowTile = blockIdx.x * 4 + wave;       // 16-row tile id
  const int row0 = rowTile * 16;
  if (row0 >= N) return;                           // wave-uniform exit: EXEC all-1s below

  const int l16 = lane & 15;
  const int hi  = lane >> 4;                       // 0 or 1
  const int kb0 = hi * 8;                          // K base for elems 0..7  (<=15 < 27)
  const int kb1 = kb0 + 16;                        // K base for elems 8..15 (16..31)
  const bool fullTile = (row0 + 16 <= N);          // N%16==0 here -> always true

  // ---- A: row (row0+l16), ISA 16-bit A layout ----
  const int row = row0 + l16;
  const int rs  = row < N ? row : N - 1;           // clamp (stores are guarded)
  const float* xr = x + (long)rs * F_IN;
  v16h a;
  #pragma unroll
  for (int j = 0; j < 8; ++j) {
    int k1 = kb1 + j;
    a[j]     = (_Float16)xr[kb0 + j];                              // kb0+j <= 15 < 27
    a[8 + j] = (k1 < F_IN) ? (_Float16)xr[k1] : (_Float16)0.f;     // 16..31 vs 27
  }

  #pragma unroll
  for (int ct = 0; ct < 4; ++ct) {
    const int col = ct * 16 + l16;
    // ---- B: column col of W1 (27x64 row-major), same K striping ----
    const float* wc = W1 + col;
    v16h b;
    #pragma unroll
    for (int j = 0; j < 8; ++j) {
      int k1 = kb1 + j;
      b[j]     = (_Float16)wc[(kb0 + j) * F1];
      b[8 + j] = (k1 < F_IN) ? (_Float16)wc[k1 * F1] : (_Float16)0.f;
    }
    v8f c = {};
    c = __builtin_amdgcn_wmma_f32_16x16x32_f16(false, a, false, b,
                                               (short)0, c, false, false);
    // ---- D layout: VGPR v -> row (v + 8*hi), lane l16 -> col ----
    float* dst = h1 + (long)(row0 + hi * 8) * F1 + col;
    if (fullTile) {
      #pragma unroll
      for (int v = 0; v < 8; ++v) dst[v * F1] = c[v];   // 8 plain b32 stores, imm offsets
    } else {
      #pragma unroll
      for (int v = 0; v < 8; ++v)
        if (row0 + hi * 8 + v < N) dst[v * F1] = c[v];
    }
  }
}

// ---------------------------------------------------------------------------
// Kernel 2: per-node attention logits for layer 1 + state init
// ---------------------------------------------------------------------------
__global__ void k_alpha1(const float* __restrict__ h1,
                         const float* __restrict__ a_src,
                         const float* __restrict__ a_dst,
                         float* __restrict__ asrc, float* __restrict__ adst,
                         float* __restrict__ m, float* __restrict__ den,
                         float* __restrict__ acc, int N) {
  int n = blockIdx.x * blockDim.x + threadIdx.x;
  if (n >= N) return;
  const float* hr = h1 + (long)n * F1;
  #pragma unroll
  for (int h = 0; h < NH1; ++h) {
    float as = 0.f, ad = 0.f;
    #pragma unroll
    for (int c = 0; c < NC1; ++c) {
      float v = hr[h * NC1 + c];
      as += v * a_src[h * NC1 + c];
      ad += v * a_dst[h * NC1 + c];
    }
    asrc[n * NH1 + h] = as;
    adst[n * NH1 + h] = ad;
    m[n * NH1 + h]   = __int_as_float(0xFF800000);   // -inf
    den[n * NH1 + h] = 0.f;
  }
  #pragma unroll
  for (int i = 0; i < F1; ++i) acc[(long)n * F1 + i] = 0.f;
}

// ---------------------------------------------------------------------------
// Kernels 3-5: layer-1 edge passes (segment max, segment sum, aggregate)
// ---------------------------------------------------------------------------
__global__ void k_edge1_max(const int* __restrict__ ei, int E, int N,
                            const float* __restrict__ asrc,
                            const float* __restrict__ adst,
                            float* __restrict__ m) {
  int e = blockIdx.x * blockDim.x + threadIdx.x;
  if (e >= E + N) return;
  int s, d; edge_sd(ei, e, E, s, d);
  #pragma unroll
  for (int h = 0; h < NH1; ++h) {
    float le = leakyr(asrc[s * NH1 + h] + adst[d * NH1 + h]);
    atomicMaxF(&m[d * NH1 + h], le);
  }
}

__global__ void k_edge1_sum(const int* __restrict__ ei, int E, int N,
                            const float* __restrict__ asrc,
                            const float* __restrict__ adst,
                            const float* __restrict__ m,
                            float* __restrict__ den) {
  int e = blockIdx.x * blockDim.x + threadIdx.x;
  if (e >= E + N) return;
  int s, d; edge_sd(ei, e, E, s, d);
  #pragma unroll
  for (int h = 0; h < NH1; ++h) {
    float le = leakyr(asrc[s * NH1 + h] + adst[d * NH1 + h]);
    atomicAdd(&den[d * NH1 + h], __expf(le - m[d * NH1 + h]));
  }
}

__global__ void k_edge1_agg(const int* __restrict__ ei, int E, int N,
                            const float* __restrict__ asrc,
                            const float* __restrict__ adst,
                            const float* __restrict__ m,
                            const float* __restrict__ h1,
                            float* __restrict__ acc) {
  long t = (long)blockIdx.x * blockDim.x + threadIdx.x;
  long total = (long)(E + N) * NH1;
  if (t >= total) return;
  int e = (int)(t >> 3);
  int h = (int)(t & 7);
  int s, d; edge_sd(ei, e, E, s, d);
  float le = leakyr(asrc[s * NH1 + h] + adst[d * NH1 + h]);
  float ex = __expf(le - m[d * NH1 + h]);
  const float* hs = h1 + (long)s * F1 + h * NC1;
  float* ad = acc + (long)d * F1 + h * NC1;
  #pragma unroll
  for (int c = 0; c < NC1; ++c) atomicAdd(&ad[c], hs[c] * ex);
}

// ---------------------------------------------------------------------------
// Kernel 6: finish layer 1 (divide, +bias, ELU) and fused layer-2 setup:
// h2 = out1 @ W2, layer-2 logits, and layer-2 state init
// ---------------------------------------------------------------------------
__global__ void k_finish1(float* __restrict__ acc,            // in: sums, out: out1
                          const float* __restrict__ den,
                          const float* __restrict__ b1,
                          const float* __restrict__ W2,       // [64,2]
                          const float* __restrict__ a_src2,   // [2]
                          const float* __restrict__ a_dst2,   // [2]
                          float* __restrict__ h2,
                          float* __restrict__ asrc2, float* __restrict__ adst2,
                          float* __restrict__ m2, float* __restrict__ den2,
                          float* __restrict__ acc2, int N) {
  int n = blockIdx.x * blockDim.x + threadIdx.x;
  if (n >= N) return;
  float* row = acc + (long)n * F1;
  float o0 = 0.f, o1 = 0.f;
  #pragma unroll
  for (int h = 0; h < NH1; ++h) {
    float inv = 1.f / den[n * NH1 + h];
    #pragma unroll
    for (int c = 0; c < NC1; ++c) {
      int i = h * NC1 + c;
      float z = row[i] * inv + b1[i];
      z = z > 0.f ? z : (__expf(z) - 1.f);        // ELU
      row[i] = z;
      o0 += z * W2[i * NC2 + 0];
      o1 += z * W2[i * NC2 + 1];
    }
  }
  h2[n * 2 + 0] = o0;
  h2[n * 2 + 1] = o1;
  asrc2[n] = o0 * a_src2[0] + o1 * a_src2[1];
  adst2[n] = o0 * a_dst2[0] + o1 * a_dst2[1];
  m2[n]   = __int_as_float(0xFF800000);
  den2[n] = 0.f;
  acc2[n * 2 + 0] = 0.f;
  acc2[n * 2 + 1] = 0.f;
}

// ---------------------------------------------------------------------------
// Kernels 7-9: layer-2 edge passes (single head, 2 channels)
// ---------------------------------------------------------------------------
__global__ void k_edge2_max(const int* __restrict__ ei, int E, int N,
                            const float* __restrict__ asrc2,
                            const float* __restrict__ adst2,
                            float* __restrict__ m2) {
  int e = blockIdx.x * blockDim.x + threadIdx.x;
  if (e >= E + N) return;
  int s, d; edge_sd(ei, e, E, s, d);
  atomicMaxF(&m2[d], leakyr(asrc2[s] + adst2[d]));
}

__global__ void k_edge2_sum(const int* __restrict__ ei, int E, int N,
                            const float* __restrict__ asrc2,
                            const float* __restrict__ adst2,
                            const float* __restrict__ m2,
                            float* __restrict__ den2) {
  int e = blockIdx.x * blockDim.x + threadIdx.x;
  if (e >= E + N) return;
  int s, d; edge_sd(ei, e, E, s, d);
  float le = leakyr(asrc2[s] + adst2[d]);
  atomicAdd(&den2[d], __expf(le - m2[d]));
}

__global__ void k_edge2_agg(const int* __restrict__ ei, int E, int N,
                            const float* __restrict__ asrc2,
                            const float* __restrict__ adst2,
                            const float* __restrict__ m2,
                            const float* __restrict__ h2,
                            float* __restrict__ acc2) {
  int e = blockIdx.x * blockDim.x + threadIdx.x;
  if (e >= E + N) return;
  int s, d; edge_sd(ei, e, E, s, d);
  float le = leakyr(asrc2[s] + adst2[d]);
  float ex = __expf(le - m2[d]);
  atomicAdd(&acc2[d * 2 + 0], h2[s * 2 + 0] * ex);
  atomicAdd(&acc2[d * 2 + 1], h2[s * 2 + 1] * ex);
}

// ---------------------------------------------------------------------------
// Kernel 10: finish layer 2 + log_softmax over 2 classes
// ---------------------------------------------------------------------------
__global__ void k_finish2(const float* __restrict__ acc2,
                          const float* __restrict__ den2,
                          const float* __restrict__ b2,
                          float* __restrict__ out, int N) {
  int n = blockIdx.x * blockDim.x + threadIdx.x;
  if (n >= N) return;
  float inv = 1.f / den2[n];
  float z0 = acc2[n * 2 + 0] * inv + b2[0];
  float z1 = acc2[n * 2 + 1] * inv + b2[1];
  float mx = fmaxf(z0, z1);
  float lse = mx + __logf(__expf(z0 - mx) + __expf(z1 - mx));
  out[n * 2 + 0] = z0 - lse;
  out[n * 2 + 1] = z1 - lse;
}

// ---------------------------------------------------------------------------
extern "C" void kernel_launch(void* const* d_in, const int* in_sizes, int n_in,
                              void* d_out, int out_size, void* d_ws, size_t ws_size,
                              hipStream_t stream) {
  const float* x      = (const float*)d_in[0];
  const int*   ei     = (const int*)  d_in[1];
  const float* W1     = (const float*)d_in[2];
  const float* a_src1 = (const float*)d_in[3];
  const float* a_dst1 = (const float*)d_in[4];
  const float* b1     = (const float*)d_in[5];
  const float* W2     = (const float*)d_in[6];
  const float* a_src2 = (const float*)d_in[7];
  const float* a_dst2 = (const float*)d_in[8];
  const float* b2     = (const float*)d_in[9];
  float* out = (float*)d_out;

  const int N  = in_sizes[0] / F_IN;
  const int E  = in_sizes[1] / 2;
  const int EE = E + N;

  // workspace layout (floats)
  float* ws    = (float*)d_ws;
  float* h1    = ws;                 ws += (size_t)N * F1;
  float* asrc1 = ws;                 ws += (size_t)N * NH1;
  float* adst1 = ws;                 ws += (size_t)N * NH1;
  float* m1    = ws;                 ws += (size_t)N * NH1;
  float* den1  = ws;                 ws += (size_t)N * NH1;
  float* acc1  = ws;                 ws += (size_t)N * F1;   // becomes out1
  float* h2    = ws;                 ws += (size_t)N * 2;
  float* asrc2 = ws;                 ws += (size_t)N;
  float* adst2 = ws;                 ws += (size_t)N;
  float* m2    = ws;                 ws += (size_t)N;
  float* den2  = ws;                 ws += (size_t)N;
  float* acc2  = ws;                 ws += (size_t)N * 2;

  const int T = 256;
  const int nodeBlocks = (N + T - 1) / T;
  const int edgeBlocks = (EE + T - 1) / T;
  const long aggT = (long)EE * NH1;
  const int  aggBlocks = (int)((aggT + T - 1) / T);
  const int  gemmBlocks = ((N + 15) / 16 + 3) / 4;   // 4 waves/block, 16 rows/wave

  k_gemm1<<<gemmBlocks, 128, 0, stream>>>(x, W1, h1, N);
  k_alpha1<<<nodeBlocks, T, 0, stream>>>(h1, a_src1, a_dst1,
                                         asrc1, adst1, m1, den1, acc1, N);
  k_edge1_max<<<edgeBlocks, T, 0, stream>>>(ei, E, N, asrc1, adst1, m1);
  k_edge1_sum<<<edgeBlocks, T, 0, stream>>>(ei, E, N, asrc1, adst1, m1, den1);
  k_edge1_agg<<<aggBlocks, T, 0, stream>>>(ei, E, N, asrc1, adst1, m1, h1, acc1);
  k_finish1<<<nodeBlocks, T, 0, stream>>>(acc1, den1, b1, W2, a_src2, a_dst2,
                                          h2, asrc2, adst2, m2, den2, acc2, N);
  k_edge2_max<<<edgeBlocks, T, 0, stream>>>(ei, E, N, asrc2, adst2, m2);
  k_edge2_sum<<<edgeBlocks, T, 0, stream>>>(ei, E, N, asrc2, adst2, m2, den2);
  k_edge2_agg<<<edgeBlocks, T, 0, stream>>>(ei, E, N, asrc2, adst2, m2, h2, acc2);
  k_finish2<<<nodeBlocks, T, 0, stream>>>(acc2, den2, b2, out, N);
}
